// MultiHead2DAttention_v2_62199716381038
// MI455X (gfx1250) — matrix operand order repaired and checked by reading, not compiled, not run on hardware
//
#include <hip/hip_runtime.h>
#include <hip/hip_bf16.h>

typedef __bf16 bf16_t;
typedef __attribute__((ext_vector_type(16))) __bf16 v16bf;
typedef __attribute__((ext_vector_type(8)))  __bf16 v8bf;
typedef __attribute__((ext_vector_type(8)))  float  v8f;
typedef __attribute__((ext_vector_type(4)))  float  v4f;

// Problem constants (from reference)
#define BATCH   32
#define SEQ     8
#define HH      32
#define WW      32
#define CC      128
#define NHEADS  8
#define TKD     512      // total key depth
#define QKIN    8192     // (H/4)*(W/4)*C
#define DKH     64       // TKD / NHEADS
#define DVH     16384    // H*W*VAL_FILTERS / NHEADS
#define KCONV   1152     // 3*3*128
#define IMGPIX  1024     // 32*32
#define IMGELEM 131072   // 32*32*128

// ---------------------------------------------------------------------------
// avg-pool 4x4 (stride 4) + cast to bf16.  src: (nimg,32,32,128) f32
// dst: (nimg, 8,8,128) bf16 flat.  total = nimg*8192
// ---------------------------------------------------------------------------
__global__ void pool_avg_bf16(const float* __restrict__ src,
                              bf16_t* __restrict__ dst, int total) {
  int gid = blockIdx.x * blockDim.x + threadIdx.x;
  if (gid >= total) return;
  int c   = gid & 127;
  int px  = (gid >> 7) & 7;
  int py  = (gid >> 10) & 7;
  int img = gid >> 13;
  const float* p = src + ((long)img * IMGPIX + (py * 4) * WW + px * 4) * CC + c;
  float s = 0.f;
#pragma unroll
  for (int i = 0; i < 4; ++i)
#pragma unroll
    for (int j = 0; j < 4; ++j)
      s += p[(i * WW + j) * CC];
  dst[gid] = (bf16_t)(s * 0.0625f);
}

// ---------------------------------------------------------------------------
// wT[n*K + k] = (bf16) w[k*N + n]   (weight transpose + downconvert)
// ---------------------------------------------------------------------------
__global__ void transpose_cast_bf16(const float* __restrict__ w,
                                    bf16_t* __restrict__ wT, int K, int N) {
  long gid = (long)blockIdx.x * blockDim.x + threadIdx.x;
  long total = (long)K * N;
  if (gid >= total) return;
  int k = (int)(gid % K);
  int n = (int)(gid / K);
  wT[gid] = (bf16_t)w[(long)k * N + n];
}

// ---------------------------------------------------------------------------
// plain f32 -> bf16 cast, 4 elements / thread (n multiple of 4)
// ---------------------------------------------------------------------------
__global__ void cast_f32_bf16(const float* __restrict__ src,
                              bf16_t* __restrict__ dst, long n) {
  long gid = ((long)blockIdx.x * blockDim.x + threadIdx.x) * 4;
  if (gid >= n) return;
  v4f v = *(const v4f*)(src + gid);
#pragma unroll
  for (int i = 0; i < 4; ++i) dst[gid + i] = (bf16_t)v[i];
}

// ---------------------------------------------------------------------------
// A-fragment loader for 16x16x32 bf16 WMMA (two b128 loads):
// lane l (l15=l&15, hi=l>>4) row = l15,
//   elems 0..15 = K k0+hi*16 .. +15
// ---------------------------------------------------------------------------
__device__ __forceinline__ v16bf load_afrag(const bf16_t* p, int hi8) {
  v8bf lo = *(const v8bf*)(p + 2 * hi8);
  v8bf hv = *(const v8bf*)(p + 2 * hi8 + 8);
  v16bf a;
#pragma unroll
  for (int i = 0; i < 8; ++i) { a[i] = lo[i]; a[i + 8] = hv[i]; }
  return a;
}

// ---------------------------------------------------------------------------
// WMMA bf16 GEMM:  C[M,N] = A[M,K] * Bt[N,K]^T  (f32 accumulate)
// One wave computes a 16x64 output tile (4 accumulators reuse one A frag).
// grid = (N/64, M/16), block = 32.
// B-fragment: lane l col = l15, elems 0..15 = K k0+hi*16 .. +15 (one 32B load)
// ---------------------------------------------------------------------------
__global__ void gemm_bf16_wmma_k(const bf16_t* __restrict__ A,
                                 const bf16_t* __restrict__ Bt,
                                 float* __restrict__ C, int M, int N, int K) {
  const int lane = threadIdx.x & 31;
  const int l15  = lane & 15;
  const int hi   = lane >> 4;
  const int hi8  = hi * 8;
  const int nbase = blockIdx.x * 64;
  const int mtile = blockIdx.y;
  const bf16_t* arow  = A  + (long)(mtile * 16 + l15) * K;
  const bf16_t* brow0 = Bt + (long)(nbase + l15) * K;
  v8f acc[4];
#pragma unroll
  for (int j = 0; j < 4; ++j)
#pragma unroll
    for (int r = 0; r < 8; ++r) acc[j][r] = 0.f;
  for (int k0 = 0; k0 < K; k0 += 32) {
    v16bf a = load_afrag(arow + k0, hi8);
#pragma unroll
    for (int j = 0; j < 4; ++j) {
      v16bf b = *(const v16bf*)(brow0 + (long)j * 16 * K + k0 + hi * 16);
      acc[j] = __builtin_amdgcn_wmma_f32_16x16x32_bf16(false, a, false, b,
                                                       (short)0, acc[j], false, false);
    }
  }
  const int rbase = mtile * 16 + hi * 8;
#pragma unroll
  for (int j = 0; j < 4; ++j) {
    const int col = nbase + j * 16 + l15;
#pragma unroll
    for (int r = 0; r < 8; ++r) C[(long)(rbase + r) * N + col] = acc[j][r];
  }
}

// ---------------------------------------------------------------------------
// 3x3 SAME conv as implicit GEMM with bf16 WMMA.
// img: (nimg,32,32,128) bf16; wT: (COUT,1152) bf16 with k = (ky*3+kx)*128+c.
// Each wave computes 16 pixels x 64 out-channels (4 accumulators share one
// A fragment -> 4x less A traffic).  block = 128 (4 waves):
//   COUT=128: 2 M-subtiles x 2 N-halves, grid = (1, 32, nimg)
//   COUT=64 : 4 M-subtiles x 1 N-group,  grid = (1, 16, nimg)
// ---------------------------------------------------------------------------
template <int COUT, bool OUT_BF16>
__global__ void conv3x3_wmma(const bf16_t* __restrict__ img,
                             const bf16_t* __restrict__ wT,
                             const float* __restrict__ bias,
                             void* __restrict__ outp) {
  const int lane = threadIdx.x & 31;
  const int wave = threadIdx.x >> 5;
  const int l15  = lane & 15;
  const int hi   = lane >> 4;
  const int hi8  = hi * 8;
  const int ngroups = COUT / 64;                       // 64-wide N groups
  const int nbase = (wave % ngroups) * 64;
  const int mtile = blockIdx.y * (4 / ngroups) + wave / ngroups;
  const long imgbase = (long)blockIdx.z * IMGELEM;
  const int  m = mtile * 16 + l15;
  const int  y = m >> 5, x = m & 31;
  const bf16_t* brow0 = wT + (long)(nbase + l15) * KCONV;
  v8f acc[4];
#pragma unroll
  for (int j = 0; j < 4; ++j) {
    float bv = bias[nbase + j * 16 + l15];
#pragma unroll
    for (int r = 0; r < 8; ++r) acc[j][r] = bv;
  }
#pragma unroll
  for (int pos = 0; pos < 9; ++pos) {
    const int ky = pos / 3, kx = pos % 3;
    const int iy = y + ky - 1, ix = x + kx - 1;
    const bool inb = ((unsigned)iy < 32u) && ((unsigned)ix < 32u);
    const bf16_t* apix = img + imgbase + (long)(iy * WW + ix) * CC;
#pragma unroll
    for (int cc = 0; cc < 4; ++cc) {
      const int c0 = cc * 32;
      v16bf a;
      if (inb) {
        a = load_afrag(apix + c0, hi8);
      } else {
#pragma unroll
        for (int i = 0; i < 16; ++i) a[i] = (bf16_t)0.f;   // zero padding
      }
      const int k0 = pos * CC + c0;
#pragma unroll
      for (int j = 0; j < 4; ++j) {
        v16bf b = *(const v16bf*)(brow0 + (long)j * 16 * KCONV + k0 + hi * 16);
        acc[j] = __builtin_amdgcn_wmma_f32_16x16x32_bf16(false, a, false, b,
                                                         (short)0, acc[j], false, false);
      }
    }
  }
  const int rbase = mtile * 16 + hi * 8;
#pragma unroll
  for (int j = 0; j < 4; ++j) {
    const int col = nbase + j * 16 + l15;
    if (OUT_BF16) {
      bf16_t* o = (bf16_t*)outp;
#pragma unroll
      for (int r = 0; r < 8; ++r)
        o[((long)blockIdx.z * IMGPIX + rbase + r) * COUT + col] = (bf16_t)acc[j][r];
    } else {
      float* o = (float*)outp;
#pragma unroll
      for (int r = 0; r < 8; ++r)
        o[((long)blockIdx.z * IMGPIX + rbase + r) * COUT + col] = acc[j][r];
    }
  }
}

// ---------------------------------------------------------------------------
// logits + relative-key bias + softmax.  q:(32,512) k:(256,512) f32.
// one thread per (b,h) pair; 256 threads total.
// ---------------------------------------------------------------------------
__global__ void attn_logits_softmax(const float* __restrict__ q,
                                    const float* __restrict__ k,
                                    const float* __restrict__ table_k,
                                    float* __restrict__ wts) {
  int t = blockIdx.x * blockDim.x + threadIdx.x;
  if (t >= BATCH * NHEADS) return;
  int b = t >> 3, h = t & 7;
  const float* qh = q + b * TKD + h * DKH;
  const int idxmap[SEQ] = {0, 0, 0, 0, 0, 1, 2, 3};  // clip(k-7,-3,3)+3
  float lg[SEQ], mx = -1e30f;
  for (int s = 0; s < SEQ; ++s) {
    const float* kh = k + (long)(b * SEQ + s) * TKD + h * DKH;
    const float* zk = table_k + idxmap[s] * DKH;
    float a = 0.f;
    for (int d = 0; d < DKH; ++d) a += qh[d] * (kh[d] + zk[d]);
    lg[s] = a * 0.125f;                    // q * d_kh^-0.5
    mx = fmaxf(mx, lg[s]);
  }
  float sum = 0.f;
  for (int s = 0; s < SEQ; ++s) { lg[s] = __expf(lg[s] - mx); sum += lg[s]; }
  float inv = 1.f / sum;
  for (int s = 0; s < SEQ; ++s) wts[t * SEQ + s] = lg[s] * inv;
}

// ---------------------------------------------------------------------------
// x[b,f] = sum_s w[b,h(f),s] * (vconv[b,s,f] + z_v[idxmap(s), f%16384])
// 8 contiguous elements per thread (f is 16B-aligned for the bf16 loads).
// ---------------------------------------------------------------------------
__global__ void attn_apply(const bf16_t* __restrict__ vconv,
                           const float* __restrict__ table_v,
                           const float* __restrict__ wts,
                           bf16_t* __restrict__ xout) {
  long gid  = (long)blockIdx.x * blockDim.x + threadIdx.x;
  long base = gid * 8;                       // flat over (b, 131072)
  int b = (int)(base >> 17);
  int f = (int)(base & (IMGELEM - 1));
  int h = f >> 14;                           // head = f / 16384
  int d = f & (DVH - 1);
  const float* w = wts + (b * NHEADS + h) * SEQ;
  const int idxmap[SEQ] = {0, 0, 0, 0, 0, 1, 2, 3};
  float acc[8] = {0, 0, 0, 0, 0, 0, 0, 0};
  for (int s = 0; s < SEQ; ++s) {
    float ws = w[s];
    v8bf vv = *(const v8bf*)(vconv + ((long)(b * SEQ + s) << 17) + f);
    const v4f* zp = (const v4f*)(table_v + (long)idxmap[s] * DVH + d);
    v4f z0 = zp[0], z1 = zp[1];
#pragma unroll
    for (int i = 0; i < 4; ++i) acc[i]     += ws * ((float)vv[i]     + z0[i]);
#pragma unroll
    for (int i = 0; i < 4; ++i) acc[i + 4] += ws * ((float)vv[i + 4] + z1[i]);
  }
#pragma unroll
  for (int i = 0; i < 8; ++i) xout[base + i] = (bf16_t)acc[i];
}

// ---------------------------------------------------------------------------
extern "C" void kernel_launch(void* const* d_in, const int* in_sizes, int n_in,
                              void* d_out, int out_size, void* d_ws, size_t ws_size,
                              hipStream_t stream) {
  const float* in_inputs = (const float*)d_in[0];   // (32,1,32,32,128)
  const float* in_kant   = (const float*)d_in[1];   // (32,8,32,32,128)
  const float* in_vant   = (const float*)d_in[2];   // (32,8,32,32,128)
  const float* in_wq     = (const float*)d_in[3];   // (8192,512)
  const float* in_wk     = (const float*)d_in[4];   // (8192,512)
  const float* in_wv     = (const float*)d_in[5];   // (3,3,128,128)
  const float* in_bv     = (const float*)d_in[6];   // (128,)
  const float* in_wo     = (const float*)d_in[7];   // (3,3,128,64)
  const float* in_bo     = (const float*)d_in[8];   // (64,)
  const float* in_tk     = (const float*)d_in[9];   // (8,64)
  const float* in_tv     = (const float*)d_in[10];  // (8,16384)
  float* out = (float*)d_out;                       // (32,1,32,32,64)

  char* ws = (char*)d_ws;
  auto alloc = [&](size_t bytes) {
    char* p = ws;
    ws += (bytes + 255) & ~(size_t)255;
    return p;
  };
  bf16_t* qa    = (bf16_t*)alloc((size_t)BATCH * QKIN * 2);
  bf16_t* ka    = (bf16_t*)alloc((size_t)BATCH * SEQ * QKIN * 2);
  bf16_t* wqT   = (bf16_t*)alloc((size_t)TKD * QKIN * 2);
  bf16_t* wkT   = (bf16_t*)alloc((size_t)TKD * QKIN * 2);
  bf16_t* wvT   = (bf16_t*)alloc((size_t)128 * KCONV * 2);
  bf16_t* woT   = (bf16_t*)alloc((size_t)64 * KCONV * 2);
  bf16_t* vantb = (bf16_t*)alloc((size_t)BATCH * SEQ * IMGELEM * 2);
  float*  qf    = (float*)alloc((size_t)BATCH * TKD * 4);
  float*  kf    = (float*)alloc((size_t)BATCH * SEQ * TKD * 4);
  float*  wts   = (float*)alloc((size_t)BATCH * NHEADS * SEQ * 4);
  bf16_t* vconv = (bf16_t*)alloc((size_t)BATCH * SEQ * IMGELEM * 2);
  bf16_t* xb    = (bf16_t*)alloc((size_t)BATCH * IMGELEM * 2);
  (void)in_sizes; (void)n_in; (void)out_size; (void)ws_size;

  // 1) pooled q/k inputs -> bf16 rows of length 8192
  pool_avg_bf16<<<(BATCH * QKIN) / 256, 256, 0, stream>>>(in_inputs, qa, BATCH * QKIN);
  pool_avg_bf16<<<(BATCH * SEQ * QKIN) / 256, 256, 0, stream>>>(in_kant, ka, BATCH * SEQ * QKIN);

  // 2) weights -> transposed bf16 [N][K]
  transpose_cast_bf16<<<(int)(((long)TKD * QKIN + 255) / 256), 256, 0, stream>>>(in_wq, wqT, QKIN, TKD);
  transpose_cast_bf16<<<(int)(((long)TKD * QKIN + 255) / 256), 256, 0, stream>>>(in_wk, wkT, QKIN, TKD);
  transpose_cast_bf16<<<(128 * KCONV + 255) / 256, 256, 0, stream>>>(in_wv, wvT, KCONV, 128);
  transpose_cast_bf16<<<(64 * KCONV + 255) / 256, 256, 0, stream>>>(in_wo, woT, KCONV, 64);

  // 3) v_antecedent -> bf16
  cast_f32_bf16<<<(int)(((long)BATCH * SEQ * IMGELEM / 4) / 256), 256, 0, stream>>>(
      in_vant, vantb, (long)BATCH * SEQ * IMGELEM);

  // 4) q = qa @ wq (32x512, K=8192); k = ka @ wk (256x512, K=8192)  [WMMA]
  gemm_bf16_wmma_k<<<dim3(TKD / 64, BATCH / 16), 32, 0, stream>>>(qa, wqT, qf, BATCH, TKD, QKIN);
  gemm_bf16_wmma_k<<<dim3(TKD / 64, (BATCH * SEQ) / 16), 32, 0, stream>>>(ka, wkT, kf, BATCH * SEQ, TKD, QKIN);

  // 5) conv_v: implicit GEMM, M=256*1024, N=128, K=1152  [WMMA] -> bf16
  conv3x3_wmma<128, true><<<dim3(1, IMGPIX / 16 / 2, BATCH * SEQ), 128, 0, stream>>>(
      vantb, wvT, in_bv, (void*)vconv);

  // 6) attention logits + softmax (tiny)
  attn_logits_softmax<<<1, 256, 0, stream>>>(qf, kf, in_tk, wts);

  // 7) weighted value sum + relative-v embedding -> x (bf16)
  attn_apply<<<(int)(((long)BATCH * IMGELEM / 8) / 256), 256, 0, stream>>>(
      vconv, in_tv, wts, xb);

  // 8) conv_o: implicit GEMM, M=32*1024, N=64, K=1152  [WMMA] -> f32 out
  conv3x3_wmma<64, false><<<dim3(1, IMGPIX / 16 / 4, BATCH), 128, 0, stream>>>(
      xb, woT, in_bo, (void*)out);
}
